// BertAttention_33758442947415
// MI455X (gfx1250) — matrix operand order repaired and checked by reading, not compiled, not run on hardware
//
#include <hip/hip_runtime.h>
#include <hip/hip_bf16.h>

// ---------------------------------------------------------------------------
// BERT attention, fused flash-style, bf16 WMMA (v_wmma_f32_16x16x32_bf16)
// B=2, S=2048, HID=1024, NH=16, HD=64, NB=32, MAXD=128
// Double-buffered fragment pipelines to overlap VMEM with WMMA.
// ---------------------------------------------------------------------------

typedef __attribute__((ext_vector_type(16))) __bf16 v16bf;
typedef __attribute__((ext_vector_type(8)))  float  v8f;

#define Bsz 2
#define Ssz 2048
#define HID 1024
#define NH 16
#define HD 64

union Frag16 {
    v16bf v;
    uint4 q[2];
    unsigned short h[16];
};

__device__ __forceinline__ unsigned short f2bf(float x) {
    unsigned u = __float_as_uint(x);
    unsigned r = u + 0x7FFFu + ((u >> 16) & 1u);   // round-to-nearest-even
    return (unsigned short)(r >> 16);
}

// A fragment, 16x32 bf16, row-major source with leading dim ld (elements).
// lane<16: row=lane,   K chunks {0..7} and {16..23}
// lane>=16: row=lane-16, K chunks {8..15} and {24..31}
__device__ __forceinline__ Frag16 load_a_frag(const unsigned short* base, int ld, int lane) {
    Frag16 f;
    int half = lane >> 4, r = lane & 15;
    const unsigned short* p = base + r * ld + half * 8;
    f.q[0] = *(const uint4*)(p);
    f.q[1] = *(const uint4*)(p + 16);
    return f;
}

// B fragment, 32x16 bf16, column data contiguous in K with leading dim ld.
// lane<16: col=lane, K chunks {0..7} and {16..23} ; lane>=16: col=lane-16, K chunks {8..15} and {24..31}
__device__ __forceinline__ Frag16 load_b_frag(const unsigned short* base, int ld, int lane) {
    Frag16 f;
    int half = lane >> 4, c = lane & 15;
    const unsigned short* p = base + c * ld + half * 8;
    f.q[0] = *(const uint4*)(p);
    f.q[1] = *(const uint4*)(p + 16);
    return f;
}

__device__ __forceinline__ v8f wmma_bf16(const Frag16& a, const Frag16& b, v8f c) {
    return __builtin_amdgcn_wmma_f32_16x16x32_bf16(false, a.v, false, b.v, (short)0, c, false, false);
}

// ---------------------------------------------------------------------------
// Conversion kernels
// ---------------------------------------------------------------------------
__global__ void cvt_bf16(const float* __restrict__ in, unsigned short* __restrict__ out, int n) {
    int i = blockIdx.x * blockDim.x + threadIdx.x;
    if (i < n) out[i] = f2bf(in[i]);
}

// in: [K][N] f32 row-major  ->  out: [N][K] bf16 (transposed)
__global__ void cvt_bf16_T(const float* __restrict__ in, unsigned short* __restrict__ out, int K, int N) {
    int i = blockIdx.x * blockDim.x + threadIdx.x;
    if (i < K * N) {
        int n = i / K, k = i - n * K;
        out[i] = f2bf(in[k * N + n]);
    }
}

// ---------------------------------------------------------------------------
// Double-buffered 32x64 GEMM core: acc[mi][c] += A[row0+16mi, :] * WT[n0+16c, :]^T
// ---------------------------------------------------------------------------
__device__ __forceinline__ void gemm_core_32x64(const unsigned short* __restrict__ A,
                                                const unsigned short* __restrict__ WT,
                                                int row0, int n0, int lane,
                                                v8f acc[2][4]) {
    Frag16 a[2][2], b[2][4];
    #pragma unroll
    for (int mi = 0; mi < 2; ++mi)
        a[0][mi] = load_a_frag(A + (row0 + 16 * mi) * HID, HID, lane);
    #pragma unroll
    for (int c = 0; c < 4; ++c)
        b[0][c] = load_b_frag(WT + (n0 + c * 16) * HID, HID, lane);

    int cur = 0;
    for (int k0 = 0; k0 < HID; k0 += 32) {
        int nxt = cur ^ 1;
        if (k0 + 32 < HID) {   // prefetch next K-slice while current WMMAs run
            #pragma unroll
            for (int mi = 0; mi < 2; ++mi)
                a[nxt][mi] = load_a_frag(A + (row0 + 16 * mi) * HID + k0 + 32, HID, lane);
            #pragma unroll
            for (int c = 0; c < 4; ++c)
                b[nxt][c] = load_b_frag(WT + (n0 + c * 16) * HID + k0 + 32, HID, lane);
        }
        #pragma unroll
        for (int mi = 0; mi < 2; ++mi)
            #pragma unroll
            for (int c = 0; c < 4; ++c)
                acc[mi][c] = wmma_bf16(a[cur][mi], b[cur][c], acc[mi][c]);
        cur = nxt;
    }
}

// ---------------------------------------------------------------------------
// QKV GEMM: [4096,1024]bf16 @ [1024,3072] (+bias) -> Q,K [b,h,s,d], V^T [b,h,d,s]
// one wave computes a 32x64 tile; grid (128, 48), block 32
// ---------------------------------------------------------------------------
__global__ void qkv_gemm(const unsigned short* __restrict__ A,
                         const unsigned short* __restrict__ WT,   // [3072][1024] bf16
                         const float* __restrict__ bias,          // [3072]
                         unsigned short* __restrict__ qb,
                         unsigned short* __restrict__ kb,
                         unsigned short* __restrict__ vt) {
    int lane = threadIdx.x;
    int row0 = blockIdx.x * 32;
    int n0   = blockIdx.y * 64;
    v8f acc[2][4] = {};
    gemm_core_32x64(A, WT, row0, n0, lane, acc);

    int half = lane >> 4, l16 = lane & 15;
    #pragma unroll
    for (int c = 0; c < 4; ++c) {
        int n = n0 + c * 16 + l16;
        float bn = bias[n];
        int sec = n >> 10;            // 0=Q 1=K 2=V (uniform within this c-block)
        int within = n & 1023;
        int h = within >> 6, d = within & 63;
        #pragma unroll
        for (int mi = 0; mi < 2; ++mi) {
            #pragma unroll
            for (int j = 0; j < 8; ++j) {
                int row = row0 + 16 * mi + j + half * 8;
                int b = row >> 11, s = row & 2047;
                int bh = b * NH + h;
                unsigned short hv = f2bf(acc[mi][c][j] + bn);
                if (sec == 0)      qb[((bh * Ssz + s) << 6) + d] = hv;
                else if (sec == 1) kb[((bh * Ssz + s) << 6) + d] = hv;
                else               vt[((bh * HD + d) << 11) + s] = hv;
            }
        }
    }
}

// ---------------------------------------------------------------------------
// Fused attention with online softmax.
// One wave per (b*h, 16-row q tile). grid (128, 32), block 32.
// ---------------------------------------------------------------------------
__device__ __forceinline__ float relbias(const float* __restrict__ table, int h, int q, int k) {
    int rel = k - q;
    int bucket = (rel > 0) ? 16 : 0;
    int a = rel < 0 ? -rel : rel;
    if (a < 8) {
        bucket += a;
    } else {
        // 8 + floor( log(a/8) / log(16) * 8 ), clamped to 15
        float t = __logf((float)a * 0.125f) * (8.0f / 2.77258872f);
        int lg = 8 + (int)t;
        bucket += (lg > 15 ? 15 : lg);
    }
    return table[h * 32 + bucket];
}

__global__ void attn_fused(const unsigned short* __restrict__ qb,
                           const unsigned short* __restrict__ kb,
                           const unsigned short* __restrict__ vt,
                           const float* __restrict__ table,
                           unsigned short* __restrict__ ctx) {  // [b][s][HID] bf16
    __shared__ __align__(32) unsigned short pbuf[16 * 32];
    int lane = threadIdx.x;
    int q0 = blockIdx.x * 16;
    int bh = blockIdx.y;
    int h  = bh & (NH - 1);
    int b  = bh >> 4;
    int half = lane >> 4, l16 = lane & 15;

    const unsigned short* Q  = qb + (bh * Ssz + q0) * HD;
    const unsigned short* Kb = kb + bh * Ssz * HD;
    const unsigned short* Vb = vt + bh * HD * Ssz;
    Frag16 qa0 = load_a_frag(Q + 0,  HD, lane);
    Frag16 qa1 = load_a_frag(Q + 32, HD, lane);

    v8f o[4] = {};
    float m[8], l[8];
    #pragma unroll
    for (int j = 0; j < 8; ++j) { m[j] = -1e30f; l[j] = 0.f; }

    // K-tile fragment double buffer (two 16-col score tiles x two d-chunks)
    Frag16 kf[2][4], vf[4];
    {
        const unsigned short* K0 = Kb;
        kf[0][0] = load_b_frag(K0 + 0,            HD, lane);
        kf[0][1] = load_b_frag(K0 + 32,           HD, lane);
        kf[0][2] = load_b_frag(K0 + 16 * HD + 0,  HD, lane);
        kf[0][3] = load_b_frag(K0 + 16 * HD + 32, HD, lane);
    }

    int cur = 0;
    for (int kt = 0; kt < Ssz; kt += 32) {
        int nxt = cur ^ 1;
        // V-tile fragments needed only after softmax: issue loads now (overlap)
        #pragma unroll
        for (int c = 0; c < 4; ++c)
            vf[c] = load_b_frag(Vb + (c * 16) * Ssz + kt, Ssz, lane);

        // ---- scores: two 16x16 tiles covering keys kt..kt+31 ----
        v8f s0 = {}, s1 = {};
        s0 = wmma_bf16(qa0, kf[cur][0], s0);
        s0 = wmma_bf16(qa1, kf[cur][1], s0);
        s1 = wmma_bf16(qa0, kf[cur][2], s1);
        s1 = wmma_bf16(qa1, kf[cur][3], s1);

        // prefetch next K-tile fragments (overlaps the softmax VALU section)
        if (kt + 32 < Ssz) {
            const unsigned short* K0 = Kb + (kt + 32) * HD;
            kf[nxt][0] = load_b_frag(K0 + 0,            HD, lane);
            kf[nxt][1] = load_b_frag(K0 + 32,           HD, lane);
            kf[nxt][2] = load_b_frag(K0 + 16 * HD + 0,  HD, lane);
            kf[nxt][3] = load_b_frag(K0 + 16 * HD + 32, HD, lane);
        }

        // ---- scale + relative position bias + online softmax ----
        #pragma unroll
        for (int j = 0; j < 8; ++j) {
            int qrow = q0 + j + half * 8;
            int kc   = kt + l16;
            s0[j] = s0[j] * 0.125f + relbias(table, h, qrow, kc);
            s1[j] = s1[j] * 0.125f + relbias(table, h, qrow, kc + 16);

            float mx = fmaxf(s0[j], s1[j]);
            mx = fmaxf(mx, __shfl_xor(mx, 1, 32));
            mx = fmaxf(mx, __shfl_xor(mx, 2, 32));
            mx = fmaxf(mx, __shfl_xor(mx, 4, 32));
            mx = fmaxf(mx, __shfl_xor(mx, 8, 32));   // per-row max within 16-lane half
            float mnew  = fmaxf(m[j], mx);
            float alpha = __expf(m[j] - mnew);
            m[j] = mnew;
            float p0 = __expf(s0[j] - mnew);
            float p1 = __expf(s1[j] - mnew);
            s0[j] = p0; s1[j] = p1;
            float rs = p0 + p1;
            rs += __shfl_xor(rs, 1, 32);
            rs += __shfl_xor(rs, 2, 32);
            rs += __shfl_xor(rs, 4, 32);
            rs += __shfl_xor(rs, 8, 32);
            l[j] = l[j] * alpha + rs;
            #pragma unroll
            for (int c = 0; c < 4; ++c) o[c][j] *= alpha;
        }

        // ---- repack P (16x32) from C-layout to A-fragment layout via LDS ----
        #pragma unroll
        for (int j = 0; j < 8; ++j) {
            int r = j + half * 8;
            pbuf[r * 32 + l16]      = f2bf(s0[j]);
            pbuf[r * 32 + 16 + l16] = f2bf(s1[j]);
        }
        __syncthreads();   // single-wave WG: cheap, keeps DS ordering explicit
        Frag16 pa;
        {
            const unsigned short* p = pbuf + l16 * 32 + half * 8;
            pa.q[0] = *(const uint4*)(p);
            pa.q[1] = *(const uint4*)(p + 16);
        }
        __syncthreads();

        // ---- O += P @ V  (V fragments already in flight/registers) ----
        #pragma unroll
        for (int c = 0; c < 4; ++c)
            o[c] = wmma_bf16(pa, vf[c], o[c]);

        cur = nxt;
    }
    // ---- epilogue: normalize and store ctx[b][q][h*64+d] as bf16 ----
    #pragma unroll
    for (int c = 0; c < 4; ++c) {
        int d = c * 16 + l16;
        #pragma unroll
        for (int j = 0; j < 8; ++j) {
            int qrow = q0 + j + half * 8;
            float val = o[c][j] / l[j];
            ctx[(b * Ssz + qrow) * HID + h * HD + d] = f2bf(val);
        }
    }
}

// ---------------------------------------------------------------------------
// Dense GEMM: ctx[4096,1024]bf16 @ w_dense (+bias) -> f32 out, 32x64 per wave
// ---------------------------------------------------------------------------
__global__ void dense_gemm(const unsigned short* __restrict__ A,
                           const unsigned short* __restrict__ WT,  // [1024][1024] bf16
                           const float* __restrict__ bias,
                           float* __restrict__ out) {
    int lane = threadIdx.x;
    int row0 = blockIdx.x * 32;
    int n0   = blockIdx.y * 64;
    v8f acc[2][4] = {};
    gemm_core_32x64(A, WT, row0, n0, lane, acc);

    int half = lane >> 4, l16 = lane & 15;
    #pragma unroll
    for (int c = 0; c < 4; ++c) {
        int n = n0 + c * 16 + l16;
        float bn = bias[n];
        #pragma unroll
        for (int mi = 0; mi < 2; ++mi) {
            #pragma unroll
            for (int j = 0; j < 8; ++j) {
                int row = row0 + 16 * mi + j + half * 8;
                out[row * HID + n] = acc[mi][c][j] + bn;
            }
        }
    }
}

// ---------------------------------------------------------------------------
extern "C" void kernel_launch(void* const* d_in, const int* in_sizes, int n_in,
                              void* d_out, int out_size, void* d_ws, size_t ws_size,
                              hipStream_t stream) {
    const float* hs      = (const float*)d_in[0];   // [2,2048,1024]
    const float* w_qkv   = (const float*)d_in[1];   // [1024,3072]
    const float* b_qkv   = (const float*)d_in[2];   // [3072]
    const float* w_dense = (const float*)d_in[3];   // [1024,1024]
    const float* b_dense = (const float*)d_in[4];   // [1024]
    const float* table   = (const float*)d_in[5];   // [16,32]
    float* out = (float*)d_out;

    char* ws = (char*)d_ws;
    const size_t MB = 1024 * 1024;
    unsigned short* hs_bf = (unsigned short*)(ws + 0 * MB);   // 8 MB
    unsigned short* wqkvT = (unsigned short*)(ws + 8 * MB);   // 6 MB
    unsigned short* wdT   = (unsigned short*)(ws + 14 * MB);  // 2 MB
    unsigned short* qb    = (unsigned short*)(ws + 16 * MB);  // 8 MB  [b,h,s,d]
    unsigned short* kb    = (unsigned short*)(ws + 24 * MB);  // 8 MB  [b,h,s,d]
    unsigned short* vt    = (unsigned short*)(ws + 32 * MB);  // 8 MB  [b,h,d,s]
    unsigned short* ctx   = (unsigned short*)(ws + 40 * MB);  // 8 MB  [b,s,HID]

    const int n_hs = Bsz * Ssz * HID;          // 4194304
    const int n_wq = HID * 3 * HID;            // 3145728
    const int n_wd = HID * HID;                // 1048576

    cvt_bf16  <<<(n_hs + 255) / 256, 256, 0, stream>>>(hs, hs_bf, n_hs);
    cvt_bf16_T<<<(n_wq + 255) / 256, 256, 0, stream>>>(w_qkv, wqkvT, HID, 3 * HID);
    cvt_bf16_T<<<(n_wd + 255) / 256, 256, 0, stream>>>(w_dense, wdT, HID, HID);

    qkv_gemm  <<<dim3(Bsz * Ssz / 32, 3 * HID / 64), 32, 0, stream>>>(hs_bf, wqkvT, b_qkv, qb, kb, vt);
    attn_fused<<<dim3(Ssz / 16, Bsz * NH),           32, 0, stream>>>(qb, kb, vt, table, ctx);
    dense_gemm<<<dim3(Bsz * Ssz / 32, HID / 64),     32, 0, stream>>>(ctx, wdT, b_dense, out);
}
